// Block_37873021616747
// MI455X (gfx1250) — compile-verified, matrix operand order checked by reading, not run on hardware
//
#include <hip/hip_runtime.h>

#define BB 4
#define TT 4096
#define CC 1024
#define HH 16
#define NN 64
#define MM (BB*TT)   // 16384 token rows

typedef __attribute__((ext_vector_type(16))) __bf16 v16bf;
typedef __attribute__((ext_vector_type(8)))  float  v8f;
typedef int v4i_ __attribute__((__vector_size__(4 * sizeof(int))));

struct Frag32B { uint4 lo, hi; };
union U8 { uint4 v; unsigned short s[8]; };

__device__ __forceinline__ unsigned short f2bf(float f) {
  union { float f; unsigned u; } v; v.f = f;
  unsigned r = v.u + 0x7fffu + ((v.u >> 16) & 1u);   // round-nearest-even
  return (unsigned short)(r >> 16);
}
__device__ __forceinline__ float bf2f(unsigned short h) {
  union { unsigned u; float f; } v; v.u = ((unsigned)h) << 16; return v.f;
}

// ---- gfx1250 async global->LDS copy (ASYNCcnt path), guarded w/ sync fallback ----
#if defined(__AMDGCN__) && __has_builtin(__builtin_amdgcn_global_load_async_to_lds_b128)
#define HAS_ASYNC_LDS 1
#else
#define HAS_ASYNC_LDS 0
#endif

__device__ __forceinline__ void cp_b128(const unsigned short* g, unsigned short* l) {
#if HAS_ASYNC_LDS
  __builtin_amdgcn_global_load_async_to_lds_b128(
      (__attribute__((address_space(1))) v4i_*)g,
      (__attribute__((address_space(3))) v4i_*)l, 0, 0);
#else
  *(uint4*)l = *(const uint4*)g;
#endif
}
__device__ __forceinline__ void cp_wait() {
#if HAS_ASYNC_LDS
#if __has_builtin(__builtin_amdgcn_s_wait_asynccnt)
  __builtin_amdgcn_s_wait_asynccnt(0);
#else
  asm volatile("s_wait_asynccnt 0" ::: "memory");
#endif
#endif
}

// 16-bit A/B fragment per CDNA5 ISA layout:
// lane 0-15: row = lane,   K = 0..7 and 16..23
// lane16-31: row = lane-16,K = 8..15 and 24..31
__device__ __forceinline__ v16bf ldsFragPtr(const unsigned short* p) {
  Frag32B fr;
  fr.lo = *(const uint4*)(p);        // K+0..7
  fr.hi = *(const uint4*)(p + 16);   // K+16..23
  return __builtin_bit_cast(v16bf, fr);
}
__device__ __forceinline__ v16bf fragAt(const unsigned short* tile, int row0, int k0, int ld) {
  int lane = (int)(threadIdx.x & 31);
  return ldsFragPtr(tile + (size_t)(row0 + (lane & 15)) * ld + k0 + ((lane >> 4) << 3));
}
__device__ __forceinline__ v8f wmma_bf16(v16bf a, v16bf b, v8f c) {
  return __builtin_amdgcn_wmma_f32_16x16x32_bf16(false, a, false, b, (short)0, c, false, false);
}

// ---------------- fp32 -> bf16 weight conversion ----------------
__global__ __launch_bounds__(256)
void cvt_kernel(const float* __restrict__ src, unsigned short* __restrict__ dst, int n) {
  int i = blockIdx.x * 256 + threadIdx.x;
  if (i < n) dst[i] = f2bf(src[i]);
}

// ---------------- LayerNorm over C=1024, output bf16 ----------------
__global__ __launch_bounds__(256)
void ln_bf16_kernel(const float* __restrict__ x, const float* __restrict__ w,
                    const float* __restrict__ b, unsigned short* __restrict__ out) {
  int row = blockIdx.x;
  const float* xr = x + (size_t)row * CC;
  float4 v = ((const float4*)xr)[threadIdx.x];           // 256 thr * 4 = 1024
  float s  = v.x + v.y + v.z + v.w;
  float s2 = v.x*v.x + v.y*v.y + v.z*v.z + v.w*v.w;
  #pragma unroll
  for (int o = 16; o > 0; o >>= 1) { s += __shfl_xor(s, o); s2 += __shfl_xor(s2, o); }
  __shared__ float ps[8], ps2[8];
  int wave = threadIdx.x >> 5, lane = threadIdx.x & 31;
  if (lane == 0) { ps[wave] = s; ps2[wave] = s2; }
  __syncthreads();
  if (threadIdx.x == 0) {
    float t = 0.f, t2 = 0.f;
    for (int i = 0; i < 8; ++i) { t += ps[i]; t2 += ps2[i]; }
    ps[0] = t; ps2[0] = t2;
  }
  __syncthreads();
  float mu = ps[0] * (1.f / CC);
  float var = ps2[0] * (1.f / CC) - mu * mu;
  float rs = rsqrtf(var + 1e-5f);
  int c = threadIdx.x * 4;
  unsigned short o4[4];
  o4[0] = f2bf((v.x - mu) * rs * w[c+0] + b[c+0]);
  o4[1] = f2bf((v.y - mu) * rs * w[c+1] + b[c+1]);
  o4[2] = f2bf((v.z - mu) * rs * w[c+2] + b[c+2]);
  o4[3] = f2bf((v.w - mu) * rs * w[c+3] + b[c+3]);
  *(ulonglong1*)(out + (size_t)row * CC + c) = *(ulonglong1*)o4;
}

// ---------------- WMMA GEMM: Y = epi( X(MxK,bf16) @ W(NxK,bf16)^T ) ----------------
// maa != null : X is built on the fly as  a + (shift(a) - a) * maa[k]  (RWKV token mix)
// epi: 0=bf16 store, 1=silu->f32, 2=relu^2->bf16, 3=sigmoid->f32, 4=+res->f32, 5=mul*acc+res->f32
__global__ __launch_bounds__(256)
void gemm_bf16_kernel(const unsigned short* __restrict__ X,
                      const unsigned short* __restrict__ Wt,
                      const float* __restrict__ maa,
                      const float* __restrict__ res,
                      const float* __restrict__ mul,
                      float* __restrict__ outF,
                      unsigned short* __restrict__ outB,
                      int M, int N, int K, int epi) {
  const int ld = 40;  // 32 + 8 pad (multiple of 8 -> 16B aligned fragments)
  __shared__ __align__(16) unsigned short Xs[128 * 40];
  __shared__ __align__(16) unsigned short Ws[128 * 40];
  int tid = threadIdx.x;
  int n0 = blockIdx.x * 128;
  int m0 = blockIdx.y * 128;
  int wave = tid >> 5;
  int wm = wave >> 1, wn = wave & 1;   // 4x2 wave grid, each wave 32x64
  v8f acc[2][4];
  #pragma unroll
  for (int i = 0; i < 2; ++i)
    #pragma unroll
    for (int j = 0; j < 4; ++j) acc[i][j] = (v8f){0.f,0.f,0.f,0.f,0.f,0.f,0.f,0.f};

  int r  = tid >> 1;           // 0..127 staging row
  int kc = (tid & 1) * 16;     // 0 or 16
  for (int k0 = 0; k0 < K; k0 += 32) {
    __syncthreads();
    // ---- stage X tile (with optional fused time-shift mix) ----
    {
      int m = m0 + r;
      const unsigned short* src = X + (size_t)m * K + k0 + kc;
      if (maa) {
        int t = m & (TT - 1);
        const unsigned short* prv = src - K;
        U8 a0, a1, p0, p1, o0, o1;
        a0.v = ((const uint4*)src)[0]; a1.v = ((const uint4*)src)[1];
        if (t > 0) { p0.v = ((const uint4*)prv)[0]; p1.v = ((const uint4*)prv)[1]; }
        else       { p0.v = make_uint4(0,0,0,0);    p1.v = make_uint4(0,0,0,0); }
        #pragma unroll
        for (int i = 0; i < 8; ++i) {
          float mk0 = maa[k0 + kc + i], mk1 = maa[k0 + kc + 8 + i];
          float av0 = bf2f(a0.s[i]), av1 = bf2f(a1.s[i]);
          float pv0 = (t > 0) ? bf2f(p0.s[i]) : 0.f;
          float pv1 = (t > 0) ? bf2f(p1.s[i]) : 0.f;
          o0.s[i] = f2bf(av0 + (pv0 - av0) * mk0);
          o1.s[i] = f2bf(av1 + (pv1 - av1) * mk1);
        }
        ((uint4*)(Xs + r * ld + kc))[0] = o0.v;
        ((uint4*)(Xs + r * ld + kc))[1] = o1.v;
      } else {
        cp_b128(src,     Xs + r * ld + kc);
        cp_b128(src + 8, Xs + r * ld + kc + 8);
      }
      if (k0 + 32 < K) __builtin_prefetch((const void*)(src + 32), 0, 3);
      // ---- stage W tile (async -> LDS) ----
      const unsigned short* wsrc = Wt + (size_t)(n0 + r) * K + k0 + kc;
      cp_b128(wsrc,     Ws + r * ld + kc);
      cp_b128(wsrc + 8, Ws + r * ld + kc + 8);
      if (k0 + 32 < K) __builtin_prefetch((const void*)(wsrc + 32), 0, 3);
    }
    cp_wait();
    __syncthreads();
    // ---- 8 WMMAs per wave per K-step ----
    v16bf a0 = fragAt(Xs, wm * 32,      0, ld);
    v16bf a1 = fragAt(Xs, wm * 32 + 16, 0, ld);
    #pragma unroll
    for (int ni = 0; ni < 4; ++ni) {
      v16bf b = fragAt(Ws, wn * 64 + ni * 16, 0, ld);
      acc[0][ni] = wmma_bf16(a0, b, acc[0][ni]);
      acc[1][ni] = wmma_bf16(a1, b, acc[1][ni]);
    }
  }
  // ---- epilogue; C/D layout: col = lane&15, row = j + 8*(lane>=16) ----
  int lane = tid & 31;
  int rowBase = m0 + wm * 32;
  int colBase = n0 + wn * 64;
  #pragma unroll
  for (int mi = 0; mi < 2; ++mi)
    #pragma unroll
    for (int ni = 0; ni < 4; ++ni) {
      int col = colBase + ni * 16 + (lane & 15);
      #pragma unroll
      for (int j = 0; j < 8; ++j) {
        int row = rowBase + mi * 16 + j + ((lane >> 4) << 3);
        size_t idx = (size_t)row * N + col;
        float v = acc[mi][ni][j];
        if      (epi == 0) outB[idx] = f2bf(v);
        else if (epi == 1) outF[idx] = v / (1.f + __expf(-v));                 // silu
        else if (epi == 2) { float q = fmaxf(v, 0.f); outB[idx] = f2bf(q*q); } // relu^2
        else if (epi == 3) outF[idx] = 1.f / (1.f + __expf(-v));               // sigmoid
        else if (epi == 4) outF[idx] = v + res[idx];                           // +residual
        else               outF[idx] = mul[idx] * v + res[idx];                // gate*acc+res
      }
    }
}

// ---------------- chunked WKV scan, one block per (b,h), Q=64 ----------------
__global__ __launch_bounds__(128)
void wkv_scan_kernel(const unsigned short* __restrict__ Rb,
                     const unsigned short* __restrict__ Kb,
                     const unsigned short* __restrict__ Vb,
                     const float* __restrict__ decay,
                     const float* __restrict__ faaaa,
                     float* __restrict__ Yb) {
  const int ld = 72;  // 64 + 8 pad
  __shared__ __align__(16) unsigned short Rs [64 * 72];
  __shared__ __align__(16) unsigned short Ks [64 * 72];
  __shared__ __align__(16) unsigned short VsT[64 * 72];  // V transposed: [m][t]
  __shared__ __align__(16) unsigned short KtW[64 * 72];  // k^T * w^{Q-1-t}: [n][t]
  __shared__ __align__(16) unsigned short Ab [64 * 72];  // a*Wm (bf16): [i][t]
  __shared__ __align__(16) unsigned short SbT[64 * 72];  // state^T bf16: [m][n]
  __shared__ __align__(16) float Sfp[64 * 68];           // state fp32: [n][m]
  __shared__ float wp[65];
  int bh = blockIdx.x, b = bh >> 4, h = bh & 15;
  int tid = threadIdx.x, wave = tid >> 5, lane = tid & 31;
  float w = __expf(-__expf(decay[h]));
  float u = faaaa[h];
  if (tid <= 64) wp[tid] = powf(w, (float)tid);
  for (int i = tid; i < 64 * 68; i += 128) Sfp[i] = 0.f;
  for (int i = tid; i < 64 * 72; i += 128) SbT[i] = 0;
  __syncthreads();
  size_t row0 = (size_t)b * TT;
  int hc = h * NN;
  for (int ch = 0; ch < TT / 64; ++ch) {
    // ---- stage R, K (async row-major) and V (manual transpose) ----
    {
      int r = tid >> 1, half = tid & 1;
      size_t g = (row0 + (size_t)ch * 64 + r) * CC + hc + half * 32;
      #pragma unroll
      for (int q = 0; q < 4; ++q) {
        cp_b128(Rb + g + q * 8, Rs + r * ld + half * 32 + q * 8);
        cp_b128(Kb + g + q * 8, Ks + r * ld + half * 32 + q * 8);
      }
      const uint4* vp = (const uint4*)(Vb + g);
      #pragma unroll
      for (int q = 0; q < 4; ++q) {
        U8 uv; uv.v = vp[q];
        #pragma unroll
        for (int e = 0; e < 8; ++e) VsT[(half * 32 + q * 8 + e) * ld + r] = uv.s[e];
      }
    }
    cp_wait();
    __syncthreads();
    // ---- KtW[n][t] = K[t][n] * w^{63-t} ----
    {
      int n = tid >> 1, half = tid & 1;
      #pragma unroll
      for (int j = 0; j < 32; ++j) {
        int t = half * 32 + j;
        KtW[n * ld + t] = f2bf(bf2f(Ks[t * ld + n]) * wp[63 - t]);
      }
    }
    __syncthreads();
    // ---- a = R K^T ; apply intra-chunk decay mask Wm; store bf16 ----
    {
      int i0 = wave * 16;
      for (int tj = 0; tj < 4; ++tj) {
        v8f acc = (v8f){0.f,0.f,0.f,0.f,0.f,0.f,0.f,0.f};
        #pragma unroll
        for (int kk = 0; kk < 64; kk += 32)
          acc = wmma_bf16(fragAt(Rs, i0, kk, ld), fragAt(Ks, tj * 16, kk, ld), acc);
        int jc = tj * 16 + (lane & 15);
        #pragma unroll
        for (int j = 0; j < 8; ++j) {
          int i = i0 + j + ((lane >> 4) << 3);
          float f = (i > jc) ? wp[i - jc - 1] : ((i == jc) ? u : 0.f);
          Ab[i * ld + jc] = f2bf(acc[j] * f);
        }
      }
    }
    __syncthreads();
    // ---- y = (a*Wm) @ V + (R @ S) * w^i ----
    {
      int i0 = wave * 16;
      for (int tj = 0; tj < 4; ++tj) {
        v8f a1 = (v8f){0.f,0.f,0.f,0.f,0.f,0.f,0.f,0.f};
        v8f a2 = (v8f){0.f,0.f,0.f,0.f,0.f,0.f,0.f,0.f};
        #pragma unroll
        for (int kk = 0; kk < 64; kk += 32) {
          a1 = wmma_bf16(fragAt(Ab, i0, kk, ld), fragAt(VsT, tj * 16, kk, ld), a1);
          a2 = wmma_bf16(fragAt(Rs, i0, kk, ld), fragAt(SbT, tj * 16, kk, ld), a2);
        }
        int m = tj * 16 + (lane & 15);
        #pragma unroll
        for (int j = 0; j < 8; ++j) {
          int i = i0 + j + ((lane >> 4) << 3);
          Yb[(row0 + (size_t)ch * 64 + i) * CC + hc + m] = a1[j] + a2[j] * wp[i];
        }
      }
    }
    __syncthreads();
    // ---- S = w^64 * S + (k*w) @ V ----
    {
      int n0 = wave * 16;
      float ws64 = wp[64];
      for (int tj = 0; tj < 4; ++tj) {
        v8f acc = (v8f){0.f,0.f,0.f,0.f,0.f,0.f,0.f,0.f};
        #pragma unroll
        for (int kk = 0; kk < 64; kk += 32)
          acc = wmma_bf16(fragAt(KtW, n0, kk, ld), fragAt(VsT, tj * 16, kk, ld), acc);
        int m = tj * 16 + (lane & 15);
        #pragma unroll
        for (int j = 0; j < 8; ++j) {
          int n = n0 + j + ((lane >> 4) << 3);
          float s = ws64 * Sfp[n * 68 + m] + acc[j];
          Sfp[n * 68 + m] = s;
          SbT[m * ld + n] = f2bf(s);
        }
      }
    }
    __syncthreads();
  }
}

// ---------------- per-head GroupNorm (eps=64e-5) fused with silu-gate multiply ----------------
__global__ __launch_bounds__(128)
void gnmul_kernel(const float* __restrict__ Y, const float* __restrict__ G,
                  const float* __restrict__ gw, const float* __restrict__ gb,
                  unsigned short* __restrict__ Att) {
  int row = blockIdx.x;
  int wave = threadIdx.x >> 5, lane = threadIdx.x & 31;
  for (int q = 0; q < 4; ++q) {
    int h = wave * 4 + q;
    size_t base = (size_t)row * CC + h * NN;
    float y0 = Y[base + lane], y1 = Y[base + lane + 32];
    float s = y0 + y1, s2 = y0 * y0 + y1 * y1;
    #pragma unroll
    for (int o = 16; o > 0; o >>= 1) { s += __shfl_xor(s, o); s2 += __shfl_xor(s2, o); }
    float mu = s * (1.f / 64.f);
    float var = s2 * (1.f / 64.f) - mu * mu;
    float rs = rsqrtf(var + 64e-5f);
    int c0 = h * NN + lane;
    float a0 = ((y0 - mu) * rs) * gw[c0]      + gb[c0];
    float a1 = ((y1 - mu) * rs) * gw[c0 + 32] + gb[c0 + 32];
    a0 *= G[base + lane];
    a1 *= G[base + lane + 32];
    Att[base + lane]      = f2bf(a0);
    Att[base + lane + 32] = f2bf(a1);
  }
}

extern "C" void kernel_launch(void* const* d_in, const int* in_sizes, int n_in,
                              void* d_out, int out_size, void* d_ws, size_t ws_size,
                              hipStream_t stream) {
  (void)in_sizes; (void)n_in; (void)out_size; (void)ws_size;
  const float* x        = (const float*)d_in[0];
  const float* ln1_w    = (const float*)d_in[1];
  const float* ln1_b    = (const float*)d_in[2];
  const float* tm_maa_k = (const float*)d_in[3];
  const float* tm_maa_v = (const float*)d_in[4];
  const float* tm_maa_r = (const float*)d_in[5];
  const float* tm_maa_g = (const float*)d_in[6];
  const float* tm_decay = (const float*)d_in[7];
  const float* tm_faaaa = (const float*)d_in[8];
  const float* Wr = (const float*)d_in[9];
  const float* Wk = (const float*)d_in[10];
  const float* Wv = (const float*)d_in[11];
  const float* Wg = (const float*)d_in[12];
  const float* Wo = (const float*)d_in[13];
  const float* gn_w = (const float*)d_in[14];
  const float* gn_b = (const float*)d_in[15];
  const float* ln2_w = (const float*)d_in[16];
  const float* ln2_b = (const float*)d_in[17];
  const float* cm_maa_k = (const float*)d_in[18];
  const float* cm_maa_r = (const float*)d_in[19];
  const float* Wck = (const float*)d_in[20];
  const float* Wcv = (const float*)d_in[21];
  const float* Wcr = (const float*)d_in[22];

  char* ws = (char*)d_ws;
  size_t off = 0;
  auto take = [&](size_t bytes) -> char* {
    char* p = ws + off;
    off = (off + bytes + 255) & ~(size_t)255;
    return p;
  };
  const size_t CC2 = (size_t)CC * CC;
  unsigned short* WrB  = (unsigned short*)take(CC2 * 2);
  unsigned short* WkB  = (unsigned short*)take(CC2 * 2);
  unsigned short* WvB  = (unsigned short*)take(CC2 * 2);
  unsigned short* WgB  = (unsigned short*)take(CC2 * 2);
  unsigned short* WoB  = (unsigned short*)take(CC2 * 2);
  unsigned short* WcrB = (unsigned short*)take(CC2 * 2);
  unsigned short* WckB = (unsigned short*)take(3 * CC2 * 2);
  unsigned short* WcvB = (unsigned short*)take(3 * CC2 * 2);
  const size_t MC = (size_t)MM * CC;
  unsigned short* LN1  = (unsigned short*)take(MC * 2);
  unsigned short* Rbuf = (unsigned short*)take(MC * 2);
  unsigned short* Kbuf = (unsigned short*)take(MC * 2);
  unsigned short* Vbuf = (unsigned short*)take(MC * 2);
  float*          Gb   = (float*)take(MC * 4);
  float*          Yb   = (float*)take(MC * 4);
  unsigned short* Att  = (unsigned short*)take(MC * 2);
  float*          X2   = (float*)take(MC * 4);
  unsigned short* L2   = (unsigned short*)take(MC * 2);
  unsigned short* Hb   = (unsigned short*)take(3 * MC * 2);
  float*          Sig  = (float*)take(MC * 4);

  int cb = (int)((CC2 + 255) / 256);
  cvt_kernel<<<cb, 256, 0, stream>>>(Wr, WrB, (int)CC2);
  cvt_kernel<<<cb, 256, 0, stream>>>(Wk, WkB, (int)CC2);
  cvt_kernel<<<cb, 256, 0, stream>>>(Wv, WvB, (int)CC2);
  cvt_kernel<<<cb, 256, 0, stream>>>(Wg, WgB, (int)CC2);
  cvt_kernel<<<cb, 256, 0, stream>>>(Wo, WoB, (int)CC2);
  cvt_kernel<<<cb, 256, 0, stream>>>(Wcr, WcrB, (int)CC2);
  cvt_kernel<<<3 * cb, 256, 0, stream>>>(Wck, WckB, (int)(3 * CC2));
  cvt_kernel<<<3 * cb, 256, 0, stream>>>(Wcv, WcvB, (int)(3 * CC2));

  // ---- time-mix ----
  ln_bf16_kernel<<<MM, 256, 0, stream>>>(x, ln1_w, ln1_b, LN1);
  dim3 g8(CC / 128, MM / 128), g24(3 * CC / 128, MM / 128);
  gemm_bf16_kernel<<<g8, 256, 0, stream>>>(LN1, WrB, tm_maa_r, nullptr, nullptr, nullptr, Rbuf, MM, CC, CC, 0);
  gemm_bf16_kernel<<<g8, 256, 0, stream>>>(LN1, WkB, tm_maa_k, nullptr, nullptr, nullptr, Kbuf, MM, CC, CC, 0);
  gemm_bf16_kernel<<<g8, 256, 0, stream>>>(LN1, WvB, tm_maa_v, nullptr, nullptr, nullptr, Vbuf, MM, CC, CC, 0);
  gemm_bf16_kernel<<<g8, 256, 0, stream>>>(LN1, WgB, tm_maa_g, nullptr, nullptr, Gb, nullptr, MM, CC, CC, 1);
  wkv_scan_kernel<<<BB * HH, 128, 0, stream>>>(Rbuf, Kbuf, Vbuf, tm_decay, tm_faaaa, Yb);
  gnmul_kernel<<<MM, 128, 0, stream>>>(Yb, Gb, gn_w, gn_b, Att);
  gemm_bf16_kernel<<<g8, 256, 0, stream>>>(Att, WoB, nullptr, x, nullptr, X2, nullptr, MM, CC, CC, 4);

  // ---- channel-mix ----
  ln_bf16_kernel<<<MM, 256, 0, stream>>>(X2, ln2_w, ln2_b, L2);
  gemm_bf16_kernel<<<g24, 256, 0, stream>>>(L2, WckB, cm_maa_k, nullptr, nullptr, nullptr, Hb, MM, 3 * CC, CC, 2);
  gemm_bf16_kernel<<<g8, 256, 0, stream>>>(L2, WcrB, cm_maa_r, nullptr, nullptr, Sig, nullptr, MM, CC, CC, 3);
  gemm_bf16_kernel<<<g8, 256, 0, stream>>>(Hb, WcvB, nullptr, X2, Sig, (float*)d_out, nullptr, MM, CC, 3 * CC, 5);
}
